// CharCnnCrf_36876589203578
// MI455X (gfx1250) — matrix operand order repaired and checked by reading, not compiled, not run on hardware
//
#include <hip/hip_runtime.h>
#include <hip/hip_bf16.h>

typedef _Float16 h16;
typedef __attribute__((ext_vector_type(8)))  _Float16 v8h;
typedef __attribute__((ext_vector_type(16))) _Float16 v16h;
typedef __attribute__((ext_vector_type(8)))  float    v8f;

#define NUM_TAGS 17
#define CHAR_DIM 128
#define NF       64
#define DHW      192      // 3*NF
#define HID      128
#define BB       128
#define SS       512
#define MM       (BB*SS)  // 65536 rows
#define PAD      3
#define SP       (SS + 2*PAD)
#define NEGV     (-1000000000.0f)

// ---------------- WMMA helpers (wave32, gfx1250 layouts) ----------------

__device__ inline v8f zero8() {
  v8f z;
#pragma unroll
  for (int i = 0; i < 8; ++i) z[i] = 0.f;
  return z;
}

// loop-invariant per-lane element offsets; the loop-varying part of every
// address stays wave-uniform so hot loops carry one offset VGPR.
__device__ inline int lofs_a(int lda) {
  const int l = threadIdx.x & 31;
  return (l & 15) * lda + ((l >> 4) << 3);
}
__device__ inline int lofs_b(int ldw) {
  const int l = threadIdx.x & 31;
  return (l & 15) * ldw + ((l >> 4) << 3);
}

// A fragment 16x32 f16 (uni = uniform base: tile row 0 at current k0)
__device__ inline v16h frag_a(const h16* uni, int lofs) {
  v8h lo = *(const v8h*)(uni + lofs);
  v8h hi = *(const v8h*)(uni + lofs + 16);
  v16h r;
#pragma unroll
  for (int j = 0; j < 8; ++j) { r[j] = lo[j]; r[j + 8] = hi[j]; }
  return r;
}

// B fragment from W[N][K] row-major (uni = uniform base: tile row n0, col k0)
__device__ inline v16h frag_b(const h16* uni, int lofs) {
  v8h lo = *(const v8h*)(uni + lofs);
  v8h hi = *(const v8h*)(uni + lofs + 16);
  v16h r;
#pragma unroll
  for (int j = 0; j < 8; ++j) { r[j] = lo[j]; r[j + 8] = hi[j]; }
  return r;
}

__device__ inline v8f wmma16(v16h a, v16h b, v8f c) {
  return __builtin_amdgcn_wmma_f32_16x16x32_f16(
      false, a, false, b, (short)0, c, false, false);
}

__device__ inline float sigf(float x) { return 1.f / (1.f + __expf(-x)); }

// ---------------- prep kernels ----------------

__global__ void k_cvt(const float* __restrict__ src, h16* __restrict__ dst, int n) {
  int i = blockIdx.x * blockDim.x + threadIdx.x;
  if (i < n) dst[i] = (h16)src[i];
}

// w[f][c][dk] (NF,128,k) -> wt[f][dk*128 + c]  (implicit-GEMM weight layout)
__global__ void k_pack_conv(const float* __restrict__ w, h16* __restrict__ wt, int k) {
  int i = blockIdx.x * blockDim.x + threadIdx.x;
  int n = NF * k * CHAR_DIM;
  if (i >= n) return;
  int f  = i / (k * CHAR_DIM);
  int dk = (i / CHAR_DIM) % k;
  int c  = i % CHAR_DIM;
  wt[i] = (h16)w[((size_t)f * CHAR_DIM + c) * k + dk];
}

// padded f16 embeddings: xe[b][sp][c], sp in [0,SP), zeros in pad
__global__ void k_embed(const int* __restrict__ x, const float* __restrict__ emb,
                        h16* __restrict__ xe) {
  int gid = blockIdx.x * blockDim.x + threadIdx.x;
  if (gid >= BB * SP * CHAR_DIM) return;
  int c  = gid & (CHAR_DIM - 1);
  int sp = (gid / CHAR_DIM) % SP;
  int b  = gid / (CHAR_DIM * SP);
  float v = 0.f;
  int s = sp - PAD;
  if (s >= 0 && s < SS) v = emb[(size_t)x[b * SS + s] * CHAR_DIM + c];
  xe[gid] = (h16)v;
}

// ---------------- conv as implicit GEMM (WMMA, ping-pong pipelined) ------
// one wave/block: 32 output rows x 64 filters; K = k*128 sliding window
// (contiguous thanks to [b][s][c] layout + pad). K/32 is even for k=3,5,7.
__global__ void __launch_bounds__(32)
k_conv(const h16* __restrict__ xe, const h16* __restrict__ wt,
       const float* __restrict__ cb, h16* __restrict__ hout,
       int k, int ncol_off) {
  const int m0 = blockIdx.x * 32;
  const int b = m0 / SS, s0 = m0 % SS;
  const int K = k * CHAR_DIM;
  const h16* a0 = xe + ((size_t)b * SP + (s0 + PAD - k / 2)) * CHAR_DIM;
  const h16* a1 = a0 + (size_t)16 * CHAR_DIM;
  const int la = lofs_a(CHAR_DIM), lb = lofs_b(K);

  v8f acc[8];
#pragma unroll
  for (int t = 0; t < 8; ++t) acc[t] = zero8();

  v16h aA0, aA1, bA[4];   // fragment set A
  v16h aB0, aB1, bB[4];   // fragment set B

  auto loadA = [&](v16h& x0, v16h& x1, v16h* bf, int k0) {
    x0 = frag_a(a0 + k0, la);
    x1 = frag_a(a1 + k0, la);
#pragma unroll
    for (int t = 0; t < 4; ++t) bf[t] = frag_b(wt + (size_t)(t * 16) * K + k0, lb);
  };
  auto mul = [&](v16h& x0, v16h& x1, v16h* bf) {
#pragma unroll
    for (int t = 0; t < 4; ++t) {
      acc[t]     = wmma16(x0, bf[t], acc[t]);
      acc[4 + t] = wmma16(x1, bf[t], acc[4 + t]);
    }
  };

  const int steps = K >> 5;   // even
  loadA(aA0, aA1, bA, 0);
  int i = 0;
  for (; i + 2 < steps; i += 2) {
    loadA(aB0, aB1, bB, (i + 1) << 5);
    mul(aA0, aA1, bA);
    loadA(aA0, aA1, bA, (i + 2) << 5);
    mul(aB0, aB1, bB);
  }
  loadA(aB0, aB1, bB, (i + 1) << 5);
  mul(aA0, aA1, bA);
  mul(aB0, aB1, bB);

  const int l = threadIdx.x & 31, col = l & 15, rb = (l >> 4) << 3;
#pragma unroll
  for (int g = 0; g < 2; ++g)
#pragma unroll
    for (int t = 0; t < 4; ++t) {
      int n = t * 16 + col;
      float bias = cb[n];
#pragma unroll
      for (int r = 0; r < 8; ++r) {
        float v = acc[g * 4 + t][r] + bias;
        v = v > 0.f ? v : 0.f;  // ReLU
        hout[(size_t)(m0 + g * 16 + r + rb) * DHW + ncol_off + n] = (h16)v;
      }
    }
}

// ---------------- generic GEMM (WMMA, ping-pong pipelined) ----------------
// out_f32[M][ldo] = A_f16[M][K] * W[N][K]^T + bias ; one wave/block,
// 32x64 tile, wave-uniform bases + single lane-offset VGPR. K/32 even.
__global__ void __launch_bounds__(32)
k_gemm(const h16* __restrict__ A, int K, const h16* __restrict__ W,
       const float* __restrict__ bias, float* __restrict__ out, int ldo) {
  const int m0 = blockIdx.x * 32;
  const int n0 = blockIdx.y * 64;
  const h16* a0 = A + (size_t)m0 * K;
  const h16* a1 = a0 + (size_t)16 * K;
  const h16* w0 = W + (size_t)n0 * K;
  const int la = lofs_a(K), lb = lofs_b(K);

  v8f acc[8];
#pragma unroll
  for (int t = 0; t < 8; ++t) acc[t] = zero8();

  v16h aA0, aA1, bA[4];
  v16h aB0, aB1, bB[4];

  auto loadA = [&](v16h& x0, v16h& x1, v16h* bf, int k0) {
    x0 = frag_a(a0 + k0, la);
    x1 = frag_a(a1 + k0, la);
#pragma unroll
    for (int t = 0; t < 4; ++t) bf[t] = frag_b(w0 + (size_t)(t * 16) * K + k0, lb);
  };
  auto mul = [&](v16h& x0, v16h& x1, v16h* bf) {
#pragma unroll
    for (int t = 0; t < 4; ++t) {
      acc[t]     = wmma16(x0, bf[t], acc[t]);
      acc[4 + t] = wmma16(x1, bf[t], acc[4 + t]);
    }
  };

  const int steps = K >> 5;   // even (192->6, 256->8)
  loadA(aA0, aA1, bA, 0);
  int i = 0;
  for (; i + 2 < steps; i += 2) {
    loadA(aB0, aB1, bB, (i + 1) << 5);
    mul(aA0, aA1, bA);
    loadA(aA0, aA1, bA, (i + 2) << 5);
    mul(aB0, aB1, bB);
  }
  loadA(aB0, aB1, bB, (i + 1) << 5);
  mul(aA0, aA1, bA);
  mul(aB0, aB1, bB);

  const int l = threadIdx.x & 31, col = l & 15, rb = (l >> 4) << 3;
#pragma unroll
  for (int g = 0; g < 2; ++g)
#pragma unroll
    for (int t = 0; t < 4; ++t) {
      int n = n0 + t * 16 + col;
      float bv = bias[n];
#pragma unroll
      for (int r = 0; r < 8; ++r)
        out[(size_t)(m0 + g * 16 + r + rb) * ldo + n] = acc[g * 4 + t][r] + bv;
    }
}

// ---------------- highway elementwise ----------------
__global__ void k_highway(const float* __restrict__ a, const float* __restrict__ g,
                          const h16* __restrict__ hraw, h16* __restrict__ outp, int n) {
  int i = blockIdx.x * blockDim.x + threadIdx.x;
  if (i >= n) return;
  float t  = sigf(g[i]);
  float rl = a[i] > 0.f ? a[i] : 0.f;
  outp[i] = (h16)(t * rl + (1.f - t) * (float)hraw[i]);
}

// ---------------- LSTM recurrence ----------------
// grid = (BB/16, 2 dirs), block = 256 (8 waves). Batch-sliced: no cross-WG
// sync on the 512-step serial chain. whh fragments live in VGPRs for the
// whole kernel; per-step critical path = 4 ds_load A-frags + 16 WMMAs.
// Gate inputs (xg, 32KB/step) staged to LDS with global_load_async_to_lds
// overlapping the WMMA phase (ASYNCcnt, ISA 08_async_tensor §4).
__global__ void __launch_bounds__(256, 1)
k_lstm(const float* __restrict__ xg_f, const float* __restrict__ xg_b,
       const h16* __restrict__ whh_f, const h16* __restrict__ whh_b,
       h16* __restrict__ hout) {
  __shared__ __align__(32) h16  h_lds[16 * HID];
  __shared__ __align__(16) float c_lds[16 * HID];
  __shared__ __align__(16) float z_lds[16 * 4 * HID];
  __shared__ __align__(16) float xg_lds[16 * 4 * HID];

  const int dir = blockIdx.y;
  const float* xg = dir ? xg_b : xg_f;
  const h16* whh  = dir ? whh_b : whh_f;
  const int col_off = dir * HID;
  const int b0 = blockIdx.x * 16;
  const int w = threadIdx.x >> 5;
  const int l = threadIdx.x & 31, col = l & 15, rb = (l >> 4) << 3;

  // register-resident whh fragments: [k-slice][n-tile]
  v16h bfr[4][4];
#pragma unroll
  for (int kk = 0; kk < 4; ++kk)
#pragma unroll
    for (int t = 0; t < 4; ++t)
      bfr[kk][t] = frag_b(whh + (size_t)(w * 64 + t * 16) * HID + kk * 32,
                          lofs_b(HID));

  for (int i = threadIdx.x; i < 16 * HID; i += blockDim.x) {
    h_lds[i] = (h16)0.f;
    c_lds[i] = 0.f;
  }
  __syncthreads();

  for (int step = 0; step < SS; ++step) {
    const int s = dir ? (SS - 1 - step) : step;

    // async-stage this step's gate inputs (16 rows x 512 f32) into LDS;
    // overlaps the WMMA phase below. 16B per lane per op.
    for (int cc = threadIdx.x; cc < 16 * 128; cc += 256) {
      int bb = cc >> 7, co = cc & 127;
      const float* ga = xg + ((size_t)(b0 + bb) * SS + s) * (4 * HID) + co * 4;
      unsigned ldsoff = (unsigned)(size_t)(&xg_lds[bb * (4 * HID) + co * 4]);
      asm volatile("global_load_async_to_lds_b128 %0, %1, off"
                   :: "v"(ldsoff), "v"((unsigned long long)(size_t)ga)
                   : "memory");
    }

    v8f acc[4];
#pragma unroll
    for (int t = 0; t < 4; ++t) acc[t] = zero8();
#pragma unroll
    for (int kk = 0; kk < 4; ++kk) {
      v16h a = frag_a(h_lds + kk * 32, lofs_a(HID));
#pragma unroll
      for (int t = 0; t < 4; ++t) acc[t] = wmma16(a, bfr[kk][t], acc[t]);
    }
#pragma unroll
    for (int t = 0; t < 4; ++t) {
      int n = w * 64 + t * 16 + col;
#pragma unroll
      for (int r = 0; r < 8; ++r) z_lds[(r + rb) * (4 * HID) + n] = acc[t][r];
    }

    asm volatile("s_wait_asynccnt 0x0" ::: "memory");
    __syncthreads();

    for (int idx = threadIdx.x; idx < 16 * HID; idx += 256) {
      int bb = idx >> 7, hh = idx & (HID - 1);
      size_t row = (size_t)(b0 + bb) * SS + s;
      const float* zr = z_lds + bb * (4 * HID);
      const float* xr = xg_lds + bb * (4 * HID);
      float zi = zr[hh]           + xr[hh];
      float zf = zr[HID + hh]     + xr[HID + hh];
      float zg = zr[2 * HID + hh] + xr[2 * HID + hh];
      float zo = zr[3 * HID + hh] + xr[3 * HID + hh];
      float c = sigf(zf) * c_lds[idx] + sigf(zi) * tanhf(zg);
      float h = sigf(zo) * tanhf(c);
      c_lds[idx] = c;
      h_lds[idx] = (h16)h;
      hout[row * (2 * HID) + col_off + hh] = (h16)h;
    }
    __syncthreads();
  }
}

// ---------------- classifier + mask -> emissions ----------------
// one wave per (b,s) row; lanes 0..16 each compute one tag logit (v8h dot)
__global__ void k_cls(const h16* __restrict__ out1, const h16* __restrict__ clsw,
                      const float* __restrict__ clsb, const int* __restrict__ x,
                      float* __restrict__ em) {
  int m = blockIdx.x * 8 + (threadIdx.x >> 5);
  int j = threadIdx.x & 31;
  if (m >= MM || j >= NUM_TAGS) return;
  const v8h* av = (const v8h*)(out1 + (size_t)m * (2 * HID));
  const v8h* wv = (const v8h*)(clsw + (size_t)j * (2 * HID));
  float acc = 0.f;
#pragma unroll
  for (int k = 0; k < (2 * HID) / 8; ++k) {
    v8h x8 = av[k], w8 = wv[k];
#pragma unroll
    for (int e = 0; e < 8; ++e) acc += (float)x8[e] * (float)w8[e];
  }
  acc += clsb[j];
  float mk = (x[m] != 0) ? 1.f : 0.f;
  em[(size_t)m * NUM_TAGS + j] = acc * mk + (1.f - mk) * NEGV;
}

// ---------------- CRF: one wave32 per batch element ----------------
__global__ void k_crf(const float* __restrict__ em, const int* __restrict__ tags,
                      const float* __restrict__ mask, const float* __restrict__ trans,
                      const float* __restrict__ start_t, const float* __restrict__ end_t,
                      float* __restrict__ loss) {
  const int b = blockIdx.x;
  const int j = threadIdx.x;
  const float* emb_ = em + (size_t)b * SS * NUM_TAGS;
  const int jj = (j < NUM_TAGS) ? j : 0;

  float tcol[NUM_TAGS];
#pragma unroll
  for (int i = 0; i < NUM_TAGS; ++i) tcol[i] = trans[i * NUM_TAGS + jj];

  float alpha = (j < NUM_TAGS) ? (start_t[j] + emb_[j]) : NEGV;

  for (int s = 1; s < SS; ++s) {
    float v[NUM_TAGS];
    float mx = -3.4e38f;
#pragma unroll
    for (int i = 0; i < NUM_TAGS; ++i) {
      v[i] = __shfl(alpha, i, 32) + tcol[i];
      mx = fmaxf(mx, v[i]);
    }
    float ssum = 0.f;
#pragma unroll
    for (int i = 0; i < NUM_TAGS; ++i) ssum += __expf(v[i] - mx);
    float nxt = mx + __logf(ssum) + emb_[s * NUM_TAGS + jj];
    float mi = mask[b * SS + s];
    if (j < NUM_TAGS) alpha = nxt * mi + alpha * (1.f - mi);
  }

  float w2[NUM_TAGS];
  float mx2 = -3.4e38f;
#pragma unroll
  for (int i = 0; i < NUM_TAGS; ++i) {
    w2[i] = __shfl(alpha, i, 32) + end_t[i];
    mx2 = fmaxf(mx2, w2[i]);
  }
  float ssum2 = 0.f;
#pragma unroll
  for (int i = 0; i < NUM_TAGS; ++i) ssum2 += __expf(w2[i] - mx2);
  float logZ = mx2 + __logf(ssum2);

  if (j == 0) {
    const int* tg = tags + b * SS;
    const float* mk = mask + b * SS;
    float score = start_t[tg[0]] + emb_[tg[0]];
    float msum = mk[0];
    for (int s = 1; s < SS; ++s) {
      score += (trans[tg[s - 1] * NUM_TAGS + tg[s]] + emb_[s * NUM_TAGS + tg[s]]) * mk[s];
      msum += mk[s];
    }
    int last_real = (int)msum - 1;
    int lt = tg[last_real < 0 ? 0 : last_real];
    score += end_t[lt] * (last_real >= 0 ? 1.f : 0.f);
    loss[b] = logZ - score;   // -(score - logZ)
  }
}

__global__ void k_final(const float* __restrict__ loss, float* __restrict__ out) {
  __shared__ float red[BB];
  int t = threadIdx.x;
  red[t] = loss[t];
  __syncthreads();
  for (int off = BB / 2; off > 0; off >>= 1) {
    if (t < off) red[t] += red[t + off];
    __syncthreads();
  }
  if (t == 0) out[0] = red[0] / (float)BB;
}

// ---------------- launch ----------------

extern "C" void kernel_launch(void* const* d_in, const int* in_sizes, int n_in,
                              void* d_out, int out_size, void* d_ws, size_t ws_size,
                              hipStream_t stream) {
  (void)in_sizes; (void)n_in; (void)out_size; (void)ws_size;
  const int*   x     = (const int*)  d_in[0];
  const int*   tags  = (const int*)  d_in[1];
  const float* mask  = (const float*)d_in[2];
  const float* emb   = (const float*)d_in[3];
  const float* cw3   = (const float*)d_in[4];
  const float* cb3   = (const float*)d_in[5];
  const float* cw5   = (const float*)d_in[6];
  const float* cb5   = (const float*)d_in[7];
  const float* cw7   = (const float*)d_in[8];
  const float* cb7   = (const float*)d_in[9];
  const float* hw_w  = (const float*)d_in[10];
  const float* hw_b  = (const float*)d_in[11];
  const float* hwg_w = (const float*)d_in[12];
  const float* hwg_b = (const float*)d_in[13];
  const float* wih0f = (const float*)d_in[14];
  const float* whh0f = (const float*)d_in[15];
  const float* b0f   = (const float*)d_in[16];
  const float* wih0b = (const float*)d_in[17];
  const float* whh0b = (const float*)d_in[18];
  const float* b0b   = (const float*)d_in[19];
  const float* wih1f = (const float*)d_in[20];
  const float* whh1f = (const float*)d_in[21];
  const float* b1f   = (const float*)d_in[22];
  const float* wih1b = (const float*)d_in[23];
  const float* whh1b = (const float*)d_in[24];
  const float* b1b   = (const float*)d_in[25];
  const float* cls_w = (const float*)d_in[26];
  const float* cls_b = (const float*)d_in[27];
  const float* trans = (const float*)d_in[28];
  const float* start_t = (const float*)d_in[29];
  const float* end_t   = (const float*)d_in[30];
  float* out = (float*)d_out;

  char* ws = (char*)d_ws;
  size_t off = 0;
  auto alloc = [&](size_t bytes) -> void* {
    void* p = ws + off;
    off += (bytes + 255) & ~(size_t)255;
    return p;
  };

  h16*  xe    = (h16*) alloc((size_t)BB * SP * CHAR_DIM * 2);
  h16*  hbuf  = (h16*) alloc((size_t)MM * DHW * 2);          // conv output (f16)
  h16*  hwy   = (h16*) alloc((size_t)MM * DHW * 2);          // highway output (f16)
  float* xgf  = (float*)alloc((size_t)MM * 512 * 4);         // also reused as hw_a
  float* xgb  = (float*)alloc((size_t)MM * 512 * 4);         // also reused as hwg_a
  h16*  o0    = (h16*) alloc((size_t)MM * 256 * 2);
  h16*  o1    = (h16*) alloc((size_t)MM * 256 * 2);
  float* em   = (float*)alloc((size_t)MM * NUM_TAGS * 4 + 256);
  h16*  wt3   = (h16*) alloc((size_t)NF * 3 * CHAR_DIM * 2);
  h16*  wt5   = (h16*) alloc((size_t)NF * 5 * CHAR_DIM * 2);
  h16*  wt7   = (h16*) alloc((size_t)NF * 7 * CHAR_DIM * 2);
  h16*  hw_w16  = (h16*)alloc((size_t)DHW * DHW * 2);
  h16*  hwg_w16 = (h16*)alloc((size_t)DHW * DHW * 2);
  h16*  wih0f16 = (h16*)alloc((size_t)512 * DHW * 2);
  h16*  wih0b16 = (h16*)alloc((size_t)512 * DHW * 2);
  h16*  wih1f16 = (h16*)alloc((size_t)512 * 256 * 2);
  h16*  wih1b16 = (h16*)alloc((size_t)512 * 256 * 2);
  h16*  whh0f16 = (h16*)alloc((size_t)512 * HID * 2);
  h16*  whh0b16 = (h16*)alloc((size_t)512 * HID * 2);
  h16*  whh1f16 = (h16*)alloc((size_t)512 * HID * 2);
  h16*  whh1b16 = (h16*)alloc((size_t)512 * HID * 2);
  h16*  cls16   = (h16*)alloc((size_t)NUM_TAGS * 256 * 2);
  float* loss   = (float*)alloc((size_t)BB * 4);

  // 1) weight conversions f32 -> f16
  const float* csrc[11] = {hw_w, hwg_w, wih0f, wih0b, wih1f, wih1b,
                           whh0f, whh0b, whh1f, whh1b, cls_w};
  h16* cdst[11] = {hw_w16, hwg_w16, wih0f16, wih0b16, wih1f16, wih1b16,
                   whh0f16, whh0b16, whh1f16, whh1b16, cls16};
  const int cn[11] = {DHW * DHW, DHW * DHW, 512 * DHW, 512 * DHW, 512 * 256,
                      512 * 256, 512 * HID, 512 * HID, 512 * HID, 512 * HID,
                      NUM_TAGS * 256};
  for (int i = 0; i < 11; ++i)
    k_cvt<<<(cn[i] + 255) / 256, 256, 0, stream>>>(csrc[i], cdst[i], cn[i]);

  k_pack_conv<<<(NF * 3 * CHAR_DIM + 255) / 256, 256, 0, stream>>>(cw3, wt3, 3);
  k_pack_conv<<<(NF * 5 * CHAR_DIM + 255) / 256, 256, 0, stream>>>(cw5, wt5, 5);
  k_pack_conv<<<(NF * 7 * CHAR_DIM + 255) / 256, 256, 0, stream>>>(cw7, wt7, 7);

  // 2) embeddings (padded, f16)
  {
    int n = BB * SP * CHAR_DIM;
    k_embed<<<(n + 255) / 256, 256, 0, stream>>>(x, emb, xe);
  }

  // 3) conv bank (implicit GEMM, WMMA): one wave/block, 32 rows
  k_conv<<<MM / 32, 32, 0, stream>>>(xe, wt3, cb3, hbuf, 3, 0);
  k_conv<<<MM / 32, 32, 0, stream>>>(xe, wt5, cb5, hbuf, 5, NF);
  k_conv<<<MM / 32, 32, 0, stream>>>(xe, wt7, cb7, hbuf, 7, 2 * NF);

  // 4) highway: two GEMMs (reuse xg buffers for f32 activations) + elementwise
  float* hw_a  = xgf;
  float* hwg_a = xgb;
  k_gemm<<<dim3(MM / 32, DHW / 64), 32, 0, stream>>>(hbuf, DHW, hw_w16,  hw_b,  hw_a,  DHW);
  k_gemm<<<dim3(MM / 32, DHW / 64), 32, 0, stream>>>(hbuf, DHW, hwg_w16, hwg_b, hwg_a, DHW);
  {
    int n = MM * DHW;
    k_highway<<<(n + 255) / 256, 256, 0, stream>>>(hw_a, hwg_a, hbuf, hwy, n);
  }

  // 5) LSTM layer 0: input projections (WMMA GEMM) + recurrence
  k_gemm<<<dim3(MM / 32, 512 / 64), 32, 0, stream>>>(hwy, DHW, wih0f16, b0f, xgf, 512);
  k_gemm<<<dim3(MM / 32, 512 / 64), 32, 0, stream>>>(hwy, DHW, wih0b16, b0b, xgb, 512);
  k_lstm<<<dim3(BB / 16, 2), 256, 0, stream>>>(xgf, xgb, whh0f16, whh0b16, o0);

  // 6) LSTM layer 1
  k_gemm<<<dim3(MM / 32, 512 / 64), 32, 0, stream>>>(o0, 256, wih1f16, b1f, xgf, 512);
  k_gemm<<<dim3(MM / 32, 512 / 64), 32, 0, stream>>>(o0, 256, wih1b16, b1b, xgb, 512);
  k_lstm<<<dim3(BB / 16, 2), 256, 0, stream>>>(xgf, xgb, whh1f16, whh1b16, o1);

  // 7) classifier -> emissions
  k_cls<<<MM / 8, 256, 0, stream>>>(o1, cls16, cls_b, x, em);

  // 8) CRF forward + score, 9) reduce to scalar loss
  k_crf<<<BB, 32, 0, stream>>>(em, tags, mask, trans, start_t, end_t, loss);
  k_final<<<1, BB, 0, stream>>>(loss, out);
}
